// TeacherModel_45938970198314
// MI455X (gfx1250) — matrix operand order repaired and checked by reading, not compiled, not run on hardware
//
#include <hip/hip_runtime.h>

// ---------------------------------------------------------------------------
// Types for CDNA5 WMMA (wave32, 16x16x32 bf16 -> f32 accumulate)
// ---------------------------------------------------------------------------
typedef __bf16 bf16_t;
typedef __attribute__((ext_vector_type(16))) __bf16 v16bf;
typedef __attribute__((ext_vector_type(8)))  float  v8f;

__device__ __forceinline__ bf16_t f2bf(float f) {
  unsigned u = __builtin_bit_cast(unsigned, f);
  unsigned r = u + 0x7FFFu + ((u >> 16) & 1u);   // round-to-nearest-even
  unsigned short h = (unsigned short)(r >> 16);
  return __builtin_bit_cast(bf16_t, h);
}
__device__ __forceinline__ float bf2f(bf16_t v) {
  unsigned short h = __builtin_bit_cast(unsigned short, v);
  unsigned u = ((unsigned)h) << 16;
  return __builtin_bit_cast(float, u);
}

// ---------------------------------------------------------------------------
// f32 -> bf16 conversion
// ---------------------------------------------------------------------------
__global__ void f32_to_bf16_kernel(const float* __restrict__ in,
                                   bf16_t* __restrict__ out, long n) {
  long i = (long)blockIdx.x * blockDim.x + threadIdx.x;
  if (i < n) out[i] = f2bf(in[i]);
}

// ---------------------------------------------------------------------------
// Pack OIHW f32 weights into WMMA A-fragment order (16x32 bf16 tiles).
// A-matrix 16x32 layout used here (contiguous K): lane = hi*16 + m, element e (0..15)
//   K = hi*16 + e,  M = m
// K ordering: rsmajor=0 -> k = ci*ksq + r  (generic, small Cin)
//             rsmajor=1 -> k = r*Cin + ci  (tap-major; Cin % 32 == 0 so each
//                          32-chunk has a fixed filter tap)
// Tile index = coutTile*KC + kc ; each tile = 32 lanes * 16 elems = 512 bf16.
// ---------------------------------------------------------------------------
__global__ void pack_weights_kernel(const float* __restrict__ w,
                                    bf16_t* __restrict__ apack, int Cout,
                                    int Cin, int ksq, int Kreal, int KC,
                                    int rsmajor) {
  int t = blockIdx.x * blockDim.x + threadIdx.x;
  int total = ((Cout + 15) / 16) * KC * 512;
  if (t >= total) return;
  int tile   = t >> 9;
  int within = t & 511;
  int lane   = within >> 4;
  int e      = within & 15;
  int coutTile = tile / KC;
  int kc       = tile - coutTile * KC;
  int hi = lane >> 4;
  int m  = lane & 15;
  int co = coutTile * 16 + m;
  int k  = kc * 32 + hi * 16 + e;
  float v = 0.f;
  if (co < Cout && k < Kreal) {
    long widx;
    if (rsmajor) {
      int ci = k % Cin;
      int rr = k / Cin;
      widx = ((long)co * Cin + ci) * ksq + rr;
    } else {
      widx = (long)co * Kreal + k;
    }
    v = w[widx];
  }
  apack[t] = f2bf(v);
}

// ---------------------------------------------------------------------------
// One MMA step: read NT B-fragments from LDS, MT A-fragments from global,
// issue MT*NT independent v_wmma_f32_16x16x32_bf16 (operand reuse both ways).
// B 32x16 layout: col = lane&15, K = (lane>>4)*16 + e (contiguous 32B in LDS).
// ---------------------------------------------------------------------------
template <int MT, int NT>
__device__ __forceinline__ void mma_step(v8f (&acc)[MT][NT],
                                         const bf16_t* lds,
                                         const bf16_t* __restrict__ apack,
                                         long tilesBase, int KC, int kc,
                                         int wv, int col, int hi, int lane) {
  v16bf bfrag[NT];
#pragma unroll
  for (int s = 0; s < NT; ++s)
    bfrag[s] = *(const v16bf*)&lds[(wv * 16 + col + s * 64) * 48 + hi * 16];
#pragma unroll
  for (int m = 0; m < MT; ++m) {
    const bf16_t* aptr =
        apack + ((tilesBase + m) * (long)KC + kc) * 512 + lane * 16;
    v16bf afrag = *(const v16bf*)aptr;          // 32B contiguous A fragment
    if (m == 0) __builtin_prefetch(aptr + 512, 0, 1);  // next A tile
#pragma unroll
    for (int s = 0; s < NT; ++s)
      acc[m][s] = __builtin_amdgcn_wmma_f32_16x16x32_bf16(
          false, afrag, false, bfrag[s], (short)0, acc[m][s], false, false);
  }
}

// ---------------------------------------------------------------------------
// Implicit-GEMM conv via WMMA with LDS-staged B tile.
//
// Block = 128 threads (4 waves). Output tile: (64*NT) pixels x (16*MT) couts.
// Assumes Wout % (64*NT) == 0 (met only by the Wout=384 launches) so a block's pixels share one output
// row: oh/ow come from block-scalar math, no per-lane div/mod or pixel guard.
//
// RS=true (Cin%32==0): tap-major K; each 32-chunk is one (kh,kw) tap, fill is
// a single strided channel-plane walk (1 load + 1 ds_store per element, bounds
// hoisted per tap). RS=false: generic per-element im2col decode (tiny layers).
// C/D layout: VGPR r -> M = r + (lane>>4)*8, N = lane&15.
// ---------------------------------------------------------------------------
template <int KS, int MT, int NT, bool RS>
__global__ void __launch_bounds__(128) conv_wmma_kernel(
    const bf16_t* __restrict__ in, const bf16_t* __restrict__ apack,
    const float* __restrict__ bias, float* __restrict__ outF,
    bf16_t* __restrict__ outB, int Cin, int Cout, int Hin, int Win, int Hout,
    int Wout, int stride, int pad, int Kreal, int KC, int relu) {
  constexpr int NPIX = 64 * NT;
  __shared__ bf16_t ldsB[NPIX * 48] __attribute__((aligned(32)));

  const int t    = threadIdx.x;
  const int lane = t & 31;
  const int wv   = t >> 5;
  const int hi   = lane >> 4;
  const int col  = lane & 15;
  const int HW   = Hout * Wout;
  const int nb   = blockIdx.x * NPIX;    // NPIX pixels, one output row only if Wout % NPIX == 0
  const int oh   = nb / Wout;
  const int owB  = nb - oh * Wout;
  const int b    = blockIdx.z;
  const long tilesBase = (long)blockIdx.y * MT;
  const long inB = (long)b * Cin;

  // fill-role coordinates: thread-constant pixel fn, k parity fk (NT==1 only)
  const int fn = t & (NPIX - 1);
  const int fk = (NT == 1) ? (t >> 6) : 0;
  const int ihBase = oh * stride - pad;
  const int iwFB   = (owB + fn) * stride - pad;

  v8f zero = {};
  v8f acc[MT][NT];
#pragma unroll
  for (int m = 0; m < MT; ++m)
#pragma unroll
    for (int s = 0; s < NT; ++s) acc[m][s] = zero;

  const bf16_t bzero = __builtin_bit_cast(bf16_t, (unsigned short)0);

  if constexpr (RS) {
    // ---- tap-major path: fixed (kh,kw) per K-chunk, ci = cb*32 + kl ----
    const int CB = Cin >> 5;             // Cin / 32 (exact)
    const long plane = (long)Hin * Win;
    constexpr int R = (KS == 3) ? 9 : 1;
    int kc = 0;
    for (int r = 0; r < R; ++r) {
      const int kh = (KS == 3) ? (r / 3) : 0;
      const int kw = (KS == 3) ? (r - kh * 3) : 0;
      const int ih = ihBase + kh;
      const int iw = iwFB + kw;
      const bool ok =
          ((unsigned)ih < (unsigned)Hin) & ((unsigned)iw < (unsigned)Win);
      const long rowAddr = (long)ih * Win + iw;
      for (int cb = 0; cb < CB; ++cb, ++kc) {
        __syncthreads();
        const bf16_t* src = in + (inB + cb * 32) * plane + rowAddr;
#pragma unroll
        for (int j = 0; j < 16 * NT; ++j) {
          const int kl = (NT == 2) ? j : (2 * j + fk);
          bf16_t v = bzero;
          if (ok) v = src[(long)kl * plane];   // channel-plane walk
          ldsB[fn * 48 + kl] = v;
        }
        __syncthreads();
        mma_step<MT, NT>(acc, ldsB, apack, tilesBase, KC, kc, wv, col, hi,
                         lane);
      }
    }
  } else {
    // ---- generic path: per-element im2col decode (small-Cin layers) ----
    for (int kc = 0; kc < KC; ++kc) {
      __syncthreads();
#pragma unroll
      for (int j = 0; j < 16 * NT; ++j) {
        const int kl = (NT == 2) ? j : (2 * j + fk);
        const int k  = kc * 32 + kl;
        bf16_t v = bzero;
        if (k < Kreal) {
          int ci, kh, kw;
          if (KS == 1) {
            ci = k; kh = 0; kw = 0;
          } else {
            ci = k / 9;                  // constant divide -> mul/shift
            int rr = k - ci * 9;
            kh = rr / 3;
            kw = rr - kh * 3;
          }
          const int ih = ihBase + kh;
          const int iw = iwFB + kw;
          if (((unsigned)ih < (unsigned)Hin) & ((unsigned)iw < (unsigned)Win))
            v = in[((inB + ci) * Hin + ih) * Win + iw];
        }
        ldsB[fn * 48 + kl] = v;
      }
      __syncthreads();
      mma_step<MT, NT>(acc, ldsB, apack, tilesBase, KC, kc, wv, col, hi, lane);
    }
  }

#pragma unroll
  for (int m = 0; m < MT; ++m) {
    const int coB = ((int)tilesBase + m) * 16 + hi * 8;
#pragma unroll
    for (int s = 0; s < NT; ++s) {
      const int n = nb + wv * 16 + col + s * 64;
#pragma unroll
      for (int r = 0; r < 8; ++r) {
        const int co = coB + r;
        if (co < Cout) {
          float val = acc[m][s][r] + bias[co];
          if (relu) val = fmaxf(val, 0.f);
          const long oidx = ((long)b * Cout + co) * HW + n;
          if (outF) outF[oidx] = val;
          if (outB) outB[oidx] = f2bf(val);
        }
      }
    }
  }
}

// ---------------------------------------------------------------------------
// 2x2 max pool (bf16)
// ---------------------------------------------------------------------------
__global__ void maxpool2_kernel(const bf16_t* __restrict__ in,
                                bf16_t* __restrict__ out, int C, int Hin,
                                int Win) {
  int Hout = Hin >> 1, Wout = Win >> 1;
  long total = 8L * C * Hout * Wout;
  long i = (long)blockIdx.x * blockDim.x + threadIdx.x;
  if (i >= total) return;
  int x = (int)(i % Wout);
  long t = i / Wout;
  int y = (int)(t % Hout);
  t /= Hout;
  int c = (int)(t % C);
  int b = (int)(t / C);
  const bf16_t* p = in + (((long)b * C + c) * Hin + 2 * y) * Win + 2 * x;
  float v = fmaxf(fmaxf(bf2f(p[0]), bf2f(p[1])),
                  fmaxf(bf2f(p[Win]), bf2f(p[Win + 1])));
  out[i] = f2bf(v);
}

// ---------------------------------------------------------------------------
// concat( up2(up_src), skip ) along channels (bf16)
// ---------------------------------------------------------------------------
__global__ void upcat_kernel(const bf16_t* __restrict__ up, int Cup,
                             const bf16_t* __restrict__ skip, int Cskip,
                             bf16_t* __restrict__ out, int Hout, int Wout) {
  int Ct = Cup + Cskip;
  long total = 8L * Ct * Hout * Wout;
  long i = (long)blockIdx.x * blockDim.x + threadIdx.x;
  if (i >= total) return;
  int x = (int)(i % Wout);
  long t = i / Wout;
  int y = (int)(t % Hout);
  t /= Hout;
  int c = (int)(t % Ct);
  int b = (int)(t / Ct);
  bf16_t v;
  if (c < Cup) {
    int Hh = Hout >> 1, Wh = Wout >> 1;
    v = up[(((long)b * Cup + c) * Hh + (y >> 1)) * Wh + (x >> 1)];
  } else {
    v = skip[(((long)b * Cskip + (c - Cup)) * Hout + y) * Wout + x];
  }
  out[i] = v;
}

// ---------------------------------------------------------------------------
// xp = concat([x (3ch), seg (1ch)]) in bf16, 4 channels @384x384
// ---------------------------------------------------------------------------
__global__ void build_xp_kernel(const bf16_t* __restrict__ xb,
                                const bf16_t* __restrict__ segb,
                                bf16_t* __restrict__ xp, int H, int W) {
  long total = 8L * 4 * H * W;
  long i = (long)blockIdx.x * blockDim.x + threadIdx.x;
  if (i >= total) return;
  long hw = (long)H * W;
  long p = i % hw;
  long t = i / hw;
  int c = (int)(t % 4);
  int b = (int)(t / 4);
  xp[i] = (c < 3) ? xb[((long)b * 3 + c) * hw + p] : segb[(long)b * hw + p];
}

// ---------------------------------------------------------------------------
// peak NMS mask: interior, > 0.1 and >= 4-neighbors; border = 0
// ---------------------------------------------------------------------------
__global__ void peaks_kernel(const float* __restrict__ hm,
                             float* __restrict__ out, int C, int H, int W) {
  long total = 8L * C * H * W;
  long i = (long)blockIdx.x * blockDim.x + threadIdx.x;
  if (i >= total) return;
  int x = (int)(i % W);
  int y = (int)((i / W) % H);
  float r = 0.f;
  if (y > 0 && y < H - 1 && x > 0 && x < W - 1) {
    float c0 = hm[i];
    if (c0 > 0.1f && c0 >= hm[i - W] && c0 >= hm[i + W] && c0 >= hm[i - 1] &&
        c0 >= hm[i + 1])
      r = 1.f;
  }
  out[i] = r;
}

// ---------------------------------------------------------------------------
// Host-side helpers / launch
// ---------------------------------------------------------------------------
static inline long apsz(int Cout, int Cin, int ks) {
  int Kreal = Cin * ks * ks;
  int KC = (Kreal + 31) / 32;
  return (long)((Cout + 15) / 16) * KC * 512;
}

static inline void pack_w(hipStream_t s, const float* w, bf16_t* ap, int Cout,
                          int Cin, int ks, int rsmajor) {
  int Kreal = Cin * ks * ks;
  int KC = (Kreal + 31) / 32;
  int total = ((Cout + 15) / 16) * KC * 512;
  pack_weights_kernel<<<(total + 255) / 256, 256, 0, s>>>(
      w, ap, Cout, Cin, ks * ks, Kreal, KC, rsmajor);
}

template <int KS, int MT, int NT, bool RS>
static inline void conv_t(hipStream_t s, const bf16_t* in, const bf16_t* ap,
                          const float* bias, float* outF, bf16_t* outB,
                          int Cin, int Cout, int Hin, int Win, int Hout,
                          int Wout, int stride, int pad, int relu) {
  int Kreal = Cin * KS * KS;
  int KC = (Kreal + 31) / 32;
  int tiles = (Cout + 15) / 16;
  dim3 grid(Hout * Wout / (64 * NT), (tiles + MT - 1) / MT, 8);
  conv_wmma_kernel<KS, MT, NT, RS>
      <<<grid, 128, 0, s>>>(in, ap, bias, outF, outB, Cin, Cout, Hin, Win,
                            Hout, Wout, stride, pad, Kreal, KC, relu);
}

#define EL(b, c, h, w) ((long)(b) * (c) * (h) * (w))

extern "C" void kernel_launch(void* const* d_in, const int* in_sizes, int n_in,
                              void* d_out, int out_size, void* d_ws,
                              size_t ws_size, hipStream_t stream) {
  (void)in_sizes; (void)n_in; (void)out_size; (void)ws_size;
  const float* x     = (const float*)d_in[0];
  const float* w_d1  = (const float*)d_in[1];  const float* b_d1  = (const float*)d_in[2];
  const float* w_d2  = (const float*)d_in[3];  const float* b_d2  = (const float*)d_in[4];
  const float* w_bn  = (const float*)d_in[5];  const float* b_bn  = (const float*)d_in[6];
  const float* w_u1  = (const float*)d_in[7];  const float* b_u1  = (const float*)d_in[8];
  const float* w_u2  = (const float*)d_in[9];  const float* b_u2  = (const float*)d_in[10];
  const float* w_seg = (const float*)d_in[11]; const float* b_seg = (const float*)d_in[12];
  const float* w_p1  = (const float*)d_in[13]; const float* b_p1  = (const float*)d_in[14];
  const float* w_p2  = (const float*)d_in[15]; const float* b_p2  = (const float*)d_in[16];
  const float* w_hm  = (const float*)d_in[17]; const float* b_hm  = (const float*)d_in[18];
  const float* w_paf = (const float*)d_in[19]; const float* b_paf = (const float*)d_in[20];

  // d_out layout: seg | hm | paf | peaks (f32, tuple return order)
  float* seg_out   = (float*)d_out;
  float* hm_out    = seg_out + EL(8, 1, 384, 384);
  float* paf_out   = hm_out + EL(8, 17, 96, 96);
  float* peaks_out = paf_out + EL(8, 32, 96, 96);

  // Workspace carve-up (all bf16)
  bf16_t* p = (bf16_t*)d_ws;
  bf16_t* XB   = p; p += EL(8, 3, 384, 384);
  bf16_t* D1   = p; p += EL(8, 32, 384, 384);
  bf16_t* D1P  = p; p += EL(8, 32, 192, 192);
  bf16_t* D2   = p; p += EL(8, 64, 192, 192);
  bf16_t* D2P  = p; p += EL(8, 64, 96, 96);
  bf16_t* BN   = p; p += EL(8, 128, 96, 96);
  bf16_t* CAT1 = p; p += EL(8, 192, 192, 192);
  bf16_t* U1   = p; p += EL(8, 64, 192, 192);
  bf16_t* CAT2 = p; p += EL(8, 96, 384, 384);
  bf16_t* U2   = p; p += EL(8, 32, 384, 384);
  bf16_t* SEGB = p; p += EL(8, 1, 384, 384);
  bf16_t* XP   = p; p += EL(8, 4, 384, 384);
  bf16_t* P1   = p; p += EL(8, 64, 192, 192);
  bf16_t* P2   = p; p += EL(8, 128, 96, 96);
  bf16_t* AP_D1  = p; p += apsz(32, 3, 3);
  bf16_t* AP_D2  = p; p += apsz(64, 32, 3);
  bf16_t* AP_BN  = p; p += apsz(128, 64, 3);
  bf16_t* AP_U1  = p; p += apsz(64, 192, 3);
  bf16_t* AP_U2  = p; p += apsz(32, 96, 3);
  bf16_t* AP_SEG = p; p += apsz(1, 32, 1);
  bf16_t* AP_P1  = p; p += apsz(64, 4, 3);
  bf16_t* AP_P2  = p; p += apsz(128, 64, 3);
  bf16_t* AP_HM  = p; p += apsz(17, 128, 1);
  bf16_t* AP_PAF = p; p += apsz(32, 128, 1);

  // 0) convert input + pack all weights into WMMA A-fragment order
  {
    long nx = EL(8, 3, 384, 384);
    f32_to_bf16_kernel<<<(int)((nx + 255) / 256), 256, 0, stream>>>(x, XB, nx);
  }
  pack_w(stream, w_d1, AP_D1, 32, 3, 3, 0);     // generic (Cin=3)
  pack_w(stream, w_d2, AP_D2, 64, 32, 3, 1);    // tap-major
  pack_w(stream, w_bn, AP_BN, 128, 64, 3, 1);
  pack_w(stream, w_u1, AP_U1, 64, 192, 3, 1);
  pack_w(stream, w_u2, AP_U2, 32, 96, 3, 1);
  pack_w(stream, w_seg, AP_SEG, 1, 32, 1, 1);
  pack_w(stream, w_p1, AP_P1, 64, 4, 3, 0);     // generic (Cin=4)
  pack_w(stream, w_p2, AP_P2, 128, 64, 3, 1);
  pack_w(stream, w_hm, AP_HM, 17, 128, 1, 1);
  pack_w(stream, w_paf, AP_PAF, 32, 128, 1, 1);

  // 1) encoder
  conv_t<3, 2, 2, false>(stream, XB, AP_D1, b_d1, nullptr, D1, 3, 32, 384, 384, 384, 384, 1, 1, 1);
  {
    long n = EL(8, 32, 192, 192);
    maxpool2_kernel<<<(int)((n + 255) / 256), 256, 0, stream>>>(D1, D1P, 32, 384, 384);
  }
  conv_t<3, 2, 2, true>(stream, D1P, AP_D2, b_d2, nullptr, D2, 32, 64, 192, 192, 192, 192, 1, 1, 1);
  {
    long n = EL(8, 64, 96, 96);
    maxpool2_kernel<<<(int)((n + 255) / 256), 256, 0, stream>>>(D2, D2P, 64, 192, 192);
  }
  conv_t<3, 2, 1, true>(stream, D2P, AP_BN, b_bn, nullptr, BN, 64, 128, 96, 96, 96, 96, 1, 1, 1);

  // 2) decoder
  {
    long n = EL(8, 192, 192, 192);
    upcat_kernel<<<(int)((n + 255) / 256), 256, 0, stream>>>(BN, 128, D2, 64, CAT1, 192, 192);
  }
  conv_t<3, 2, 2, true>(stream, CAT1, AP_U1, b_u1, nullptr, U1, 192, 64, 192, 192, 192, 192, 1, 1, 1);
  {
    long n = EL(8, 96, 384, 384);
    upcat_kernel<<<(int)((n + 255) / 256), 256, 0, stream>>>(U1, 64, D1, 32, CAT2, 384, 384);
  }
  conv_t<3, 2, 2, true>(stream, CAT2, AP_U2, b_u2, nullptr, U2, 96, 32, 384, 384, 384, 384, 1, 1, 1);

  // 3) seg head (f32 to d_out, bf16 copy for pose branch input)
  conv_t<1, 1, 2, true>(stream, U2, AP_SEG, b_seg, seg_out, SEGB, 32, 1, 384, 384, 384, 384, 1, 0, 0);

  // 4) pose branch (stride-2 SAME convs: pad_lo = 0, high pad via bounds check)
  {
    long n = EL(8, 4, 384, 384);
    build_xp_kernel<<<(int)((n + 255) / 256), 256, 0, stream>>>(XB, SEGB, XP, 384, 384);
  }
  conv_t<3, 2, 2, false>(stream, XP, AP_P1, b_p1, nullptr, P1, 4, 64, 384, 384, 192, 192, 2, 0, 1);
  conv_t<3, 2, 1, true>(stream, P1, AP_P2, b_p2, nullptr, P2, 64, 128, 192, 192, 96, 96, 2, 0, 1);
  conv_t<1, 2, 1, true>(stream, P2, AP_HM, b_hm, hm_out, nullptr, 128, 17, 96, 96, 96, 96, 1, 0, 0);
  conv_t<1, 2, 1, true>(stream, P2, AP_PAF, b_paf, paf_out, nullptr, 128, 32, 96, 96, 96, 96, 1, 0, 0);

  // 5) peak mask
  {
    long n = EL(8, 17, 96, 96);
    peaks_kernel<<<(int)((n + 255) / 256), 256, 0, stream>>>(hm_out, peaks_out, 17, 96, 96);
  }
}
